// GNN_Feature_Extractor_3152505995796
// MI455X (gfx1250) — matrix operand order repaired and checked by reading, not compiled, not run on hardware
//
#include <hip/hip_runtime.h>
#include <hip/hip_bf16.h>

#define N_NODES   50000
#define E_EDGES   800000
#define E_TOT     (E_EDGES + N_NODES)   // self loops appended
#define F_IN      128
#define HID       64
#define HEADS1    4
#define C1        (HEADS1 * HID)        // 256
#define NUM_GRAPHS 64
#define NEG_SLOPE 0.2f

typedef __attribute__((ext_vector_type(16))) _Float16 v16h;
typedef __attribute__((ext_vector_type(8)))  _Float16 v8h;
typedef __attribute__((ext_vector_type(8)))  float    v8f;

union V16U { v16h v; v8h h[2]; };

static inline unsigned cdiv(unsigned a, unsigned b) { return (a + b - 1) / b; }

// ---------------------------------------------------------------- utilities

__global__ void k_fill_f32(float* __restrict__ p, float val, int n) {
  int i = blockIdx.x * blockDim.x + threadIdx.x;
  if (i < n) p[i] = val;
}

__global__ void k_cvt_f16(const float* __restrict__ src, _Float16* __restrict__ dst, int n) {
  int i = blockIdx.x * blockDim.x + threadIdx.x;
  if (i < n) dst[i] = (_Float16)src[i];
}

// Pack W [K, Ncols] (row-major f32) into per-lane WMMA B fragments (f16).
// Fragment layout: Bp[((ct*KSTEPS + s)*32 + lane)*16 + i]
//   k = s*32 + (lane>=16 ? 8 : 0) + (i&7) + (i>=8 ? 16 : 0) ,  n = ct*16 + (lane&15)
__global__ void k_pack_B(const float* __restrict__ W, _Float16* __restrict__ Bp,
                         int ksteps, int ncols) {
  int p = blockIdx.x * blockDim.x + threadIdx.x;
  int total = (ncols / 16) * ksteps * 32 * 16;
  if (p >= total) return;
  int i    = p & 15;
  int lane = (p >> 4) & 31;
  int rest = p >> 9;
  int s    = rest % ksteps;
  int ct   = rest / ksteps;
  int k = s * 32 + ((lane >> 4) << 3) + (i & 7) + ((i >> 3) << 4);
  int n = ct * 16 + (lane & 15);
  Bp[p] = (_Float16)W[k * ncols + n];
}

// ---------------------------------------------------------------- WMMA GEMM
// D[M,Ncols] (f16 out) = A[M,K](f16) * Bpacked.  One 16-row tile per block,
// one or two 16-col tiles per wave.  Requires full EXEC (no divergence).
template <int CTW, int KSTEPS>
__global__ void k_gemm_wmma(const _Float16* __restrict__ A,
                            const _Float16* __restrict__ Bp,
                            _Float16* __restrict__ Dh,
                            int ldA, int ldD) {
  const int lane    = threadIdx.x & 31;
  const int wave    = threadIdx.x >> 5;
  const int rowBase = blockIdx.x * 16;
  const int m       = lane & 15;
  const int khalf   = lane >> 4;                       // 0 or 1

  const _Float16* arow = A + (size_t)(rowBase + m) * ldA + khalf * 8;

  v8f acc[CTW];
  const v8f vzero = {0.f, 0.f, 0.f, 0.f, 0.f, 0.f, 0.f, 0.f};
#pragma unroll
  for (int t = 0; t < CTW; ++t) acc[t] = vzero;

#pragma unroll
  for (int s = 0; s < KSTEPS; ++s) {
    V16U a;
    a.h[0] = *(const v8h*)(arow + s * 32);
    a.h[1] = *(const v8h*)(arow + s * 32 + 16);
    __builtin_prefetch(arow + (s + 1) * 32, 0, 1);     // global_prefetch_b8
#pragma unroll
    for (int t = 0; t < CTW; ++t) {
      const int ct = wave * CTW + t;
      const _Float16* bp = Bp + ((size_t)(ct * KSTEPS + s) * 32 + lane) * 16;
      V16U b;
      b.h[0] = *(const v8h*)(bp);
      b.h[1] = *(const v8h*)(bp + 8);
      acc[t] = __builtin_amdgcn_wmma_f32_16x16x32_f16(
          false, a.v, false, b.v, (short)0, acc[t], false, false);
    }
  }

  // C/D layout: VGPR r, lane L -> row = r + 8*(L>=16), col = L&15
  const int rOff = 8 * khalf;
#pragma unroll
  for (int t = 0; t < CTW; ++t) {
    const int n = (wave * CTW + t) * 16 + m;
#pragma unroll
    for (int r = 0; r < 8; ++r)
      Dh[(size_t)(rowBase + r + rOff) * ldD + n] = (_Float16)acc[t][r];
  }
}

// ------------------------------------------------------- per-node attention
// asrc[n*H+h] = sum_c Hh[n, h*64+c] * a_src[h*64+c]   (same for adst)
__global__ void k_alpha(const _Float16* __restrict__ Hh,
                        const float* __restrict__ a_src,
                        const float* __restrict__ a_dst,
                        float* __restrict__ asrc, float* __restrict__ adst,
                        int heads) {
  int t = blockIdx.x * blockDim.x + threadIdx.x;
  if (t >= N_NODES * heads) return;
  int n = t / heads, h = t % heads;
  const _Float16* hp = Hh + (size_t)n * heads * 64 + h * 64;
  float s1 = 0.f, s2 = 0.f;
#pragma unroll 8
  for (int c = 0; c < 64; ++c) {
    float v = (float)hp[c];
    s1 += v * a_src[h * 64 + c];
    s2 += v * a_dst[h * 64 + c];
  }
  asrc[t] = s1;
  adst[t] = s2;
}

// ------------------------------------------------------------- edge kernels

__device__ __forceinline__ void atomicMaxFloat(float* addr, float val) {
  int iv = __float_as_int(val);
  if (iv >= 0)  atomicMax((int*)addr, iv);
  else          atomicMin((unsigned int*)addr, (unsigned int)iv);
}

__device__ __forceinline__ void edge_endpoints(const int* __restrict__ ei,
                                               int e, int& s, int& d) {
  if (e < E_EDGES) { s = ei[e]; d = ei[E_EDGES + e]; }
  else             { s = d = e - E_EDGES; }           // self loop
}

template <int H>
__global__ void k_edge_logits_max(const int* __restrict__ ei,
                                  const float* __restrict__ asrc,
                                  const float* __restrict__ adst,
                                  float* __restrict__ ebuf,
                                  float* __restrict__ mbuf) {
  int e = blockIdx.x * blockDim.x + threadIdx.x;
  if (e >= E_TOT) return;
  int s, d;
  edge_endpoints(ei, e, s, d);
#pragma unroll
  for (int h = 0; h < H; ++h) {
    float x = asrc[s * H + h] + adst[d * H + h];
    float v = x > 0.f ? x : NEG_SLOPE * x;             // leaky_relu
    ebuf[(size_t)e * H + h] = v;
    atomicMaxFloat(&mbuf[d * H + h], v);
  }
}

template <int H>
__global__ void k_edge_expsum(const int* __restrict__ ei,
                              float* __restrict__ ebuf,   // in: e, out: w
                              const float* __restrict__ mbuf,
                              float* __restrict__ dbuf) {
  int e = blockIdx.x * blockDim.x + threadIdx.x;
  if (e >= E_TOT) return;
  int s, d;
  edge_endpoints(ei, e, s, d);
#pragma unroll
  for (int h = 0; h < H; ++h) {
    float w = __expf(ebuf[(size_t)e * H + h] - mbuf[d * H + h]);
    ebuf[(size_t)e * H + h] = w;
    atomicAdd(&dbuf[d * H + h], w);
  }
}

template <int H>
__global__ void k_edge_msg(const int* __restrict__ ei,
                           const float* __restrict__ wbuf,
                           const float* __restrict__ dbuf,
                           const _Float16* __restrict__ Hh,
                           float* __restrict__ agg) {
  int t = blockIdx.x * blockDim.x + threadIdx.x;
  if (t >= E_TOT * H) return;
  int e = t / H, h = t % H;
  int s, d;
  edge_endpoints(ei, e, s, d);
  float alpha = wbuf[t] / dbuf[d * H + h];
  const _Float16* hp = Hh + (size_t)s * (H * 64) + h * 64;
  float* ap = agg + (size_t)d * (H * 64) + h * 64;
#pragma unroll
  for (int c0 = 0; c0 < 64; c0 += 8) {
    v8h hv = *(const v8h*)(hp + c0);                   // global_load_b128
#pragma unroll
    for (int j = 0; j < 8; ++j)
      atomicAdd(ap + c0 + j, alpha * (float)hv[j]);    // global_atomic_add_f32
  }
}

// ------------------------------------------------------------- epilogues

__global__ void k_bias_relu_cvt(const float* __restrict__ agg,
                                const float* __restrict__ bias,
                                _Float16* __restrict__ outh,
                                int cols, int n) {
  int i = blockIdx.x * blockDim.x + threadIdx.x;
  if (i >= n) return;
  float v = agg[i] + bias[i % cols];
  outh[i] = (_Float16)(v > 0.f ? v : 0.f);
}

__global__ void k_pool(const float* __restrict__ agg2,
                       const float* __restrict__ b2,
                       const int* __restrict__ batch,
                       float* __restrict__ sums, float* __restrict__ cnt) {
  int t = blockIdx.x * blockDim.x + threadIdx.x;
  if (t >= N_NODES * HID) return;
  int n = t >> 6, c = t & 63;
  int g = batch[n];
  atomicAdd(&sums[g * HID + c], agg2[t] + b2[c]);
  if (c == 0) atomicAdd(&cnt[g], 1.0f);
}

__global__ void k_finish(const float* __restrict__ sums,
                         const float* __restrict__ cnt,
                         float* __restrict__ out) {
  int t = blockIdx.x * blockDim.x + threadIdx.x;
  if (t >= NUM_GRAPHS * HID) return;
  out[t] = sums[t] / fmaxf(cnt[t >> 6], 1.0f);
}

// ---------------------------------------------------------------- launcher

extern "C" void kernel_launch(void* const* d_in, const int* in_sizes, int n_in,
                              void* d_out, int out_size, void* d_ws, size_t ws_size,
                              hipStream_t stream) {
  const float* x   = (const float*)d_in[0];
  const int*   ei  = (const int*)  d_in[1];
  const int*   bat = (const int*)  d_in[2];
  const float* W1  = (const float*)d_in[3];
  const float* as1 = (const float*)d_in[4];
  const float* ad1 = (const float*)d_in[5];
  const float* b1  = (const float*)d_in[6];
  const float* W2  = (const float*)d_in[7];
  const float* as2 = (const float*)d_in[8];
  const float* ad2 = (const float*)d_in[9];
  const float* b2  = (const float*)d_in[10];
  float* out = (float*)d_out;

  // -------- carve workspace
  char* p = (char*)d_ws;
  auto carve = [&](size_t bytes) -> void* {
    void* r = (void*)p;
    p += (bytes + 255) & ~(size_t)255;
    return r;
  };
  _Float16* Bp1   = (_Float16*)carve((size_t)F_IN * C1 * 2);          // 64 KB
  _Float16* Bp2   = (_Float16*)carve((size_t)C1 * HID * 2);           // 32 KB
  _Float16* xh    = (_Float16*)carve((size_t)N_NODES * F_IN * 2);     // 12.8 MB
  _Float16* h1h   = (_Float16*)carve((size_t)N_NODES * C1 * 2);       // 25.6 MB (reused as x2h)
  float* asrc1    = (float*)carve((size_t)N_NODES * HEADS1 * 4);
  float* adst1    = (float*)carve((size_t)N_NODES * HEADS1 * 4);
  float* m1       = (float*)carve((size_t)N_NODES * HEADS1 * 4);
  float* d1       = (float*)carve((size_t)N_NODES * HEADS1 * 4);
  float* e1       = (float*)carve((size_t)E_TOT * HEADS1 * 4);        // 13.6 MB
  float* agg1     = (float*)carve((size_t)N_NODES * C1 * 4);          // 51.2 MB
  _Float16* h2h   = (_Float16*)carve((size_t)N_NODES * HID * 2);      // 6.4 MB
  float* asrc2    = (float*)carve((size_t)N_NODES * 4);
  float* adst2    = (float*)carve((size_t)N_NODES * 4);
  float* m2       = (float*)carve((size_t)N_NODES * 4);
  float* d2       = (float*)carve((size_t)N_NODES * 4);
  float* e2       = (float*)carve((size_t)E_TOT * 4);
  float* agg2     = (float*)carve((size_t)N_NODES * HID * 4);         // 12.8 MB
  float* sums     = (float*)carve((size_t)NUM_GRAPHS * HID * 4);
  float* cnt      = (float*)carve((size_t)NUM_GRAPHS * 4);

  const int T = 256;

  // -------- prep: pack weights, convert x, init accumulators
  k_pack_B<<<cdiv(F_IN * C1, T), T, 0, stream>>>(W1, Bp1, F_IN / 32, C1);
  k_pack_B<<<cdiv(C1 * HID, T), T, 0, stream>>>(W2, Bp2, C1 / 32, HID);
  k_cvt_f16<<<cdiv(N_NODES * F_IN, T), T, 0, stream>>>(x, xh, N_NODES * F_IN);

  k_fill_f32<<<cdiv(N_NODES * HEADS1, T), T, 0, stream>>>(m1, -1e30f, N_NODES * HEADS1);
  k_fill_f32<<<cdiv(N_NODES * HEADS1, T), T, 0, stream>>>(d1, 0.f, N_NODES * HEADS1);
  k_fill_f32<<<cdiv(N_NODES * C1, T), T, 0, stream>>>(agg1, 0.f, N_NODES * C1);
  k_fill_f32<<<cdiv(N_NODES, T), T, 0, stream>>>(m2, -1e30f, N_NODES);
  k_fill_f32<<<cdiv(N_NODES, T), T, 0, stream>>>(d2, 0.f, N_NODES);
  k_fill_f32<<<cdiv(N_NODES * HID, T), T, 0, stream>>>(agg2, 0.f, N_NODES * HID);
  k_fill_f32<<<cdiv(NUM_GRAPHS * HID, T), T, 0, stream>>>(sums, 0.f, NUM_GRAPHS * HID);
  k_fill_f32<<<1, 64, 0, stream>>>(cnt, 0.f, NUM_GRAPHS);

  // -------- layer 1: h1 = x @ W1 (WMMA), attention, softmax, aggregate
  // 3125 row tiles; 8 waves/block, 2 col-tiles/wave -> 256 cols
  k_gemm_wmma<2, F_IN / 32><<<N_NODES / 16, 256, 0, stream>>>(xh, Bp1, h1h, F_IN, C1);

  k_alpha<<<cdiv(N_NODES * HEADS1, T), T, 0, stream>>>(h1h, as1, ad1, asrc1, adst1, HEADS1);
  k_edge_logits_max<HEADS1><<<cdiv(E_TOT, T), T, 0, stream>>>(ei, asrc1, adst1, e1, m1);
  k_edge_expsum<HEADS1><<<cdiv(E_TOT, T), T, 0, stream>>>(ei, e1, m1, d1);
  k_edge_msg<HEADS1><<<cdiv(E_TOT * HEADS1, T), T, 0, stream>>>(ei, e1, d1, h1h, agg1);

  // x2 = relu(agg1 + b1) -> f16, overwriting h1h (dead after k_edge_msg)
  k_bias_relu_cvt<<<cdiv(N_NODES * C1, T), T, 0, stream>>>(agg1, b1, h1h, C1, N_NODES * C1);

  // -------- layer 2: h2 = x2 @ W2 (WMMA), attention, softmax, aggregate
  // 3125 row tiles; 4 waves/block, 1 col-tile/wave -> 64 cols
  k_gemm_wmma<1, C1 / 32><<<N_NODES / 16, 128, 0, stream>>>(h1h, Bp2, h2h, C1, HID);

  k_alpha<<<cdiv(N_NODES, T), T, 0, stream>>>(h2h, as2, ad2, asrc2, adst2, 1);
  k_edge_logits_max<1><<<cdiv(E_TOT, T), T, 0, stream>>>(ei, asrc2, adst2, e2, m2);
  k_edge_expsum<1><<<cdiv(E_TOT, T), T, 0, stream>>>(ei, e2, m2, d2);
  k_edge_msg<1><<<cdiv(E_TOT, T), T, 0, stream>>>(ei, e2, d2, h2h, agg2);

  // -------- global mean pool
  k_pool<<<cdiv(N_NODES * HID, T), T, 0, stream>>>(agg2, b2, bat, sums, cnt);
  k_finish<<<cdiv(NUM_GRAPHS * HID, T), T, 0, stream>>>(sums, cnt, out);

  (void)in_sizes; (void)n_in; (void)out_size; (void)ws_size;
}
